// PureCartesianSparseE3Conv_37967510897144
// MI455X (gfx1250) — matrix operand order repaired and checked by reading, not compiled, not run on hardware
//
#include <hip/hip_runtime.h>
#include <cstdint>
#include <cstddef>

// ---------------------------------------------------------------------------
// Types for CDNA5 WMMA (wave32, 16x16x32 f16 -> f32)
// ---------------------------------------------------------------------------
typedef __attribute__((ext_vector_type(16))) _Float16 v16h;
typedef __attribute__((ext_vector_type(8)))  float    v8f;
typedef _Float16 h2t __attribute__((ext_vector_type(2)));

union AFrag {
  v16h     v;
  h2t      h[8];
  unsigned u[8];
  uint4    q[2];
};

__device__ __forceinline__ float silu_f(float x) { return x / (1.0f + __expf(-x)); }

__device__ __forceinline__ unsigned pack2(float a, float b) {
  h2t h; h[0] = (_Float16)a; h[1] = (_Float16)b;
  return __builtin_bit_cast(unsigned, h);
}

// ---------------------------------------------------------------------------
// Problem constants
// ---------------------------------------------------------------------------
#define NB     16
#define C_OUT  16
#define CCH    8
#define NLO    48          // (LMAX+1)*C_OUT
#define KAUG   4160        // 65*64 : (64 h rows + 1 bias row) x 64 uv
#define NBLK   130         // KAUG / 32
#define FCOLS  208         // 16 + 48 + 144 output columns per edge

// ---------------------------------------------------------------------------
// Zero-fill
// ---------------------------------------------------------------------------
__global__ void zero_f(float* p, int n) {
  int i = blockIdx.x * blockDim.x + threadIdx.x;
  if (i < n) p[i] = 0.0f;
}

// ---------------------------------------------------------------------------
// Node MLP: Ai = silu(emb_table[A] @ W1 + b1) @ W2 + b2   -> [N,8] f32
// ---------------------------------------------------------------------------
__global__ void node_mlp_k(const float* __restrict__ embt, const int* __restrict__ A,
                           const float* __restrict__ W1, const float* __restrict__ b1,
                           const float* __restrict__ W2, const float* __restrict__ b2,
                           float* __restrict__ Ai, int N) {
  int n = blockIdx.x * blockDim.x + threadIdx.x;
  if (n >= N) return;
  int a = A[n];
  float e[16];
#pragma unroll
  for (int i = 0; i < 16; ++i) e[i] = embt[a * 16 + i];
  float out[8];
#pragma unroll
  for (int c = 0; c < 8; ++c) out[c] = b2[c];
  for (int o = 0; o < 64; ++o) {
    float h = b1[o];
#pragma unroll
    for (int i = 0; i < 16; ++i) h = fmaf(e[i], W1[i * 64 + o], h);
    h = silu_f(h);
#pragma unroll
    for (int c = 0; c < 8; ++c) out[c] = fmaf(h, W2[o * 8 + c], out[c]);
  }
#pragma unroll
  for (int c = 0; c < 8; ++c) Ai[(size_t)n * 8 + c] = out[c];
}

// ---------------------------------------------------------------------------
// Edge prep: geometry, t = f_src (x) x2 (f16 pairs), radial MLP -> h (f16 dup
// pairs, 65th row = 1.0 for the fb3 bias fold), edge count atomics.
// ---------------------------------------------------------------------------
__global__ void edge_prep_k(const float* __restrict__ pos, const int* __restrict__ batch,
                            const int* __restrict__ esrc, const int* __restrict__ edst,
                            const float* __restrict__ shifts, const float* __restrict__ cell,
                            const float* __restrict__ Ai,
                            const float* __restrict__ F1, const float* __restrict__ fb1,
                            const float* __restrict__ F2, const float* __restrict__ fb2,
                            float* __restrict__ cnt, float* __restrict__ ng,
                            unsigned* __restrict__ tg, unsigned* __restrict__ hdup, int E) {
  int e = blockIdx.x * blockDim.x + threadIdx.x;
  if (e >= E) return;
  int s = esrc[e], d = edst[e];
  const float* Cm = cell + (size_t)batch[s] * 9;
  float s0 = shifts[e * 3 + 0], s1 = shifts[e * 3 + 1], s2 = shifts[e * 3 + 2];
  float v[3];
#pragma unroll
  for (int j = 0; j < 3; ++j) {
    float sh = s0 * Cm[0 * 3 + j] + s1 * Cm[1 * 3 + j] + s2 * Cm[2 * 3 + j];
    v[j] = pos[(size_t)d * 3 + j] - pos[(size_t)s * 3 + j] + sh;
  }
  float len = sqrtf(v[0] * v[0] + v[1] * v[1] + v[2] * v[2]);
#pragma unroll
  for (int j = 0; j < 3; ++j) ng[(size_t)e * 4 + j] = v[j] / len;
  ng[(size_t)e * 4 + 3] = len;

  atomicAdd(cnt + d, 1.0f);

  // t = f_src (u) * x2 (v), packed as f16 pairs over v
  float fs[8], x2[8];
#pragma unroll
  for (int u = 0; u < 8; ++u) { fs[u] = Ai[(size_t)s * 8 + u]; x2[u] = Ai[(size_t)d * 8 + u]; }
#pragma unroll
  for (int p = 0; p < 32; ++p) {
    int u = p >> 2, v0 = (p & 3) * 2;
    tg[(size_t)e * 32 + p] = pack2(fs[u] * x2[v0], fs[u] * x2[v0 + 1]);
  }

  // radial gaussian basis * sqrt(NB)
  float emb[16];
  const float step = 5.0f / 17.0f;
#pragma unroll
  for (int i = 0; i < 16; ++i) {
    float diff = (len - 5.0f * (float)(i + 1) / 17.0f) / step;
    emb[i] = __expf(-diff * diff) * (4.0f / 1.12f);
  }
  float h1[64];
#pragma unroll
  for (int o = 0; o < 64; ++o) {
    float x = fb1[o];
#pragma unroll
    for (int i = 0; i < 16; ++i) x = fmaf(emb[i], F1[i * 64 + o], x);
    h1[o] = silu_f(x);
  }
  for (int o = 0; o < 64; ++o) {   // keep dynamic: h1 read with constant k only
    float x = fb2[o];
#pragma unroll
    for (int k = 0; k < 64; ++k) x = fmaf(h1[k], F2[k * 64 + o], x);
    x = silu_f(x);
    hdup[(size_t)e * 65 + o] = pack2(x, x);
  }
  hdup[(size_t)e * 65 + 64] = 0x3C003C00u;  // (1.0h, 1.0h) bias row
}

// ---------------------------------------------------------------------------
// Pre-swizzle F3 (+fb3) into per-lane WMMA B fragments, f16.
// B 16-bit 32x16, contiguous k order used here (A matches): lanes 0-15 col=l, VGPR q holds K=2q,2q+1;
// lanes 16-31 col=l-16, K=16+2q,17+2q.  Z row index Kg = k*64 + uv.
// ---------------------------------------------------------------------------
__global__ void bprep_k(const float* __restrict__ F3, const float* __restrict__ fb3,
                        unsigned* __restrict__ Bf) {
  int t = blockIdx.x * blockDim.x + threadIdx.x;
  if (t >= NBLK * 3 * 32 * 8) return;
  int q    = t & 7;
  int lane = (t >> 3) & 31;
  int j    = (t >> 8) % 3;
  int b    = t / 768;
  int col  = j * 16 + (lane & 15);
  int half = lane >> 4;
  int Kg   = 32 * b + 16 * half + 2 * q;
  float lo, hi;
  if (Kg < 4096) {
    int k = Kg >> 6, uv = Kg & 63;
    lo = F3[(size_t)k * 3072 + col * 64 + uv];
    hi = F3[(size_t)k * 3072 + col * 64 + uv + 1];
  } else {
    int uv = Kg - 4096;
    lo = fb3[col * 64 + uv];
    hi = fb3[col * 64 + uv + 1];
  }
  Bf[t] = pack2(lo, hi);
}

// ---------------------------------------------------------------------------
// Main: one wave32 per 16-edge tile.  s[16e x 48] = Z @ F3aug via WMMA
// (K=4160 -> 130 steps x 3 N-tiles = 390 v_wmma_f32_16x16x32_f16),
// A = (h (x) t) built in-register with v_pk_mul_f16, then equivariant
// epilogue scatter-adds with global_atomic_add_f32.
// ---------------------------------------------------------------------------
__global__ __launch_bounds__(32) void e3_main_k(
    const unsigned* __restrict__ hdup, const unsigned* __restrict__ tg,
    const float* __restrict__ ng, const int* __restrict__ edst,
    const unsigned* __restrict__ Bf, float* __restrict__ tot, int E) {
  __shared__ unsigned ldsH[16 * 66];                 // stride 66 -> bank-spread
  __shared__ __align__(16) unsigned ldsT[16 * 36];   // stride 36 words, 16B rows
  __shared__ float ldsN[16 * 4];
  __shared__ int   ldsD[16];

  const int lane = threadIdx.x;
  const int e0   = blockIdx.x * 16;

  for (int i = lane; i < 16 * 65; i += 32) {
    int r = i / 65, c = i - r * 65;
    int e = e0 + r;
    ldsH[r * 66 + c] = (e < E) ? hdup[(size_t)e * 65 + c] : 0u;
  }
  for (int i = lane; i < 16 * 32; i += 32) {
    int r = i >> 5, c = i & 31;
    int e = e0 + r;
    ldsT[r * 36 + c] = (e < E) ? tg[(size_t)e * 32 + c] : 0u;
  }
  if (lane < 16) {
    int e = e0 + lane;
    ldsD[lane] = (e < E) ? edst[e] : 0;
    float nx = 0.f, ny = 0.f, nz = 0.f;
    if (e < E) { nx = ng[(size_t)e * 4 + 0]; ny = ng[(size_t)e * 4 + 1]; nz = ng[(size_t)e * 4 + 2]; }
    ldsN[lane * 4 + 0] = nx; ldsN[lane * 4 + 1] = ny; ldsN[lane * 4 + 2] = nz;
  }
  __syncthreads();

  const int row  = lane & 15;   // A-operand row = edge within tile
  const int half = lane >> 4;
  v8f c0 = {0.f, 0.f, 0.f, 0.f, 0.f, 0.f, 0.f, 0.f};
  v8f c1 = c0, c2 = c0;

  const unsigned* hrow = ldsH + row * 66;
  const unsigned* trow = ldsT + row * 36;
  const uint4*    bp0  = (const uint4*)(Bf + ((size_t)lane) * 8);

  for (int b = 0; b < NBLK; ++b) {
    // ---- A fragment: z[K] = h[k] * t[uv],  K = k*64 + uv ----
    h2t hh = __builtin_bit_cast(h2t, hrow[b >> 1]);
    int p0 = ((b & 1) << 4) + (half << 3);
    const uint4* tp = (const uint4*)(trow + p0);
    uint4 tlo = tp[0];
    uint4 thi = tp[1];
    AFrag af;
    af.h[0] = hh * __builtin_bit_cast(h2t, tlo.x);
    af.h[1] = hh * __builtin_bit_cast(h2t, tlo.y);
    af.h[2] = hh * __builtin_bit_cast(h2t, tlo.z);
    af.h[3] = hh * __builtin_bit_cast(h2t, tlo.w);
    af.h[4] = hh * __builtin_bit_cast(h2t, thi.x);
    af.h[5] = hh * __builtin_bit_cast(h2t, thi.y);
    af.h[6] = hh * __builtin_bit_cast(h2t, thi.z);
    af.h[7] = hh * __builtin_bit_cast(h2t, thi.w);

    // ---- B fragments (pre-swizzled), b128 loads; prefetch next K-step ----
    const uint4* bp = bp0 + (size_t)b * 192;  // 3 tiles * 32 lanes * 2 uint4
    AFrag bf0, bf1, bf2;
    bf0.q[0] = bp[0];   bf0.q[1] = bp[1];
    bf1.q[0] = bp[64];  bf1.q[1] = bp[65];
    bf2.q[0] = bp[128]; bf2.q[1] = bp[129];
    if (b < NBLK - 1) __builtin_prefetch(bp + 192, 0, 1);

    c0 = __builtin_amdgcn_wmma_f32_16x16x32_f16(false, af.v, false, bf0.v, (short)0, c0, false, false);
    c1 = __builtin_amdgcn_wmma_f32_16x16x32_f16(false, af.v, false, bf1.v, (short)0, c1, false, false);
    c2 = __builtin_amdgcn_wmma_f32_16x16x32_f16(false, af.v, false, bf2.v, (short)0, c2, false, false);
  }

  // ---- Equivariant epilogue + scatter-add ----
  // C layout: VGPR r, lanes 0-15 -> edge row r, col lane; lanes 16-31 -> row r+8.
  const int o = lane & 15;
#pragma unroll
  for (int r = 0; r < 8; ++r) {
    int mrow = r + (half << 3);
    int e = e0 + mrow;
    if (e < E) {
      float* bt = tot + (size_t)ldsD[mrow] * FCOLS;
      float nx = ldsN[mrow * 4 + 0];
      float ny = ldsN[mrow * 4 + 1];
      float nz = ldsN[mrow * 4 + 2];
      atomicAdd(bt + o, c0[r]);
      float s1 = c1[r];
      atomicAdd(bt + 16 + o * 3 + 0, s1 * nx);
      atomicAdd(bt + 16 + o * 3 + 1, s1 * ny);
      atomicAdd(bt + 16 + o * 3 + 2, s1 * nz);
      float s2 = c2[r];
      float na[3] = {nx, ny, nz};
#pragma unroll
      for (int ai = 0; ai < 3; ++ai)
#pragma unroll
        for (int bi = 0; bi < 3; ++bi)
          atomicAdd(bt + 64 + o * 9 + ai * 3 + bi, s2 * na[ai] * na[bi]);
    }
  }
}

// ---------------------------------------------------------------------------
// Scatter-mean finalize
// ---------------------------------------------------------------------------
__global__ void finalize_k(const float* __restrict__ tot, const float* __restrict__ cnt,
                           float* __restrict__ out, int N) {
  int i = blockIdx.x * blockDim.x + threadIdx.x;
  if (i < N * FCOLS) out[i] = tot[i] / fmaxf(cnt[i / FCOLS], 1.0f);
}

// ---------------------------------------------------------------------------
// Host launcher
// ---------------------------------------------------------------------------
extern "C" void kernel_launch(void* const* d_in, const int* in_sizes, int n_in,
                              void* d_out, int out_size, void* d_ws, size_t ws_size,
                              hipStream_t stream) {
  const float* pos    = (const float*)d_in[0];
  const int*   Aatom  = (const int*)d_in[1];
  const int*   batch  = (const int*)d_in[2];
  const int*   esrc   = (const int*)d_in[3];
  const int*   edst   = (const int*)d_in[4];
  const float* shifts = (const float*)d_in[5];
  const float* cell   = (const float*)d_in[6];
  const float* embt   = (const float*)d_in[7];
  const float* W1  = (const float*)d_in[8];
  const float* b1  = (const float*)d_in[9];
  const float* W2  = (const float*)d_in[10];
  const float* b2  = (const float*)d_in[11];
  const float* F1  = (const float*)d_in[12];
  const float* fb1 = (const float*)d_in[13];
  const float* F2  = (const float*)d_in[14];
  const float* fb2 = (const float*)d_in[15];
  const float* F3  = (const float*)d_in[16];
  const float* fb3 = (const float*)d_in[17];

  const int N = in_sizes[0] / 3;
  const int E = in_sizes[3];

  float* ws = (float*)d_ws;
  size_t off = 0;
  float*    Ai   = ws + off; off += (size_t)N * 8;
  float*    ng   = ws + off; off += (size_t)E * 4;
  unsigned* hdup = (unsigned*)(ws + off); off += (size_t)E * 65;
  unsigned* tg   = (unsigned*)(ws + off); off += (size_t)E * 32;
  unsigned* Bf   = (unsigned*)(ws + off); off += (size_t)NBLK * 3 * 32 * 8;
  float*    tot  = ws + off; off += (size_t)N * FCOLS;
  float*    cnt  = ws + off; off += (size_t)N;
  (void)ws_size; (void)n_in;

  int ztot = N * FCOLS + N;  // tot and cnt are contiguous
  zero_f<<<(ztot + 255) / 256, 256, 0, stream>>>(tot, ztot);
  node_mlp_k<<<(N + 127) / 128, 128, 0, stream>>>(embt, Aatom, W1, b1, W2, b2, Ai, N);
  bprep_k<<<(NBLK * 768 + 255) / 256, 256, 0, stream>>>(F3, fb3, Bf);
  edge_prep_k<<<(E + 63) / 64, 64, 0, stream>>>(pos, batch, esrc, edst, shifts, cell, Ai,
                                                F1, fb1, F2, fb2, cnt, ng, tg, hdup, E);
  int tiles = (E + 15) / 16;
  e3_main_k<<<tiles, 32, 0, stream>>>(hdup, tg, ng, edst, Bf, tot, E);
  finalize_k<<<(N * FCOLS + 255) / 256, 256, 0, stream>>>(tot, cnt, (float*)d_out, N);
}
